// CFOP_7825430413989
// MI455X (gfx1250) — matrix operand order repaired and checked by reading, not compiled, not run on hardware
//
#include <hip/hip_runtime.h>
#include <math.h>

// ---------------- problem constants (from reference) ----------------
constexpr int N_NODES = 200000;
constexpr int N_EDGES = 400000;
constexpr int N_GRAPH = 128;
constexpr int IN_DIM  = 512;
constexpr int HID     = 128;
constexpr int OUTD    = 64;
constexpr int N_MASK  = 100000;

typedef __attribute__((ext_vector_type(16))) __bf16 v16bf;
typedef __attribute__((ext_vector_type(8)))  float  v8f;

// ---------------- degree / dinv / mask flags ----------------
__global__ void k_degree(const int* __restrict__ src, const int* __restrict__ dst,
                         float* __restrict__ degTd, float* __restrict__ degBu, int nE) {
  int e = blockIdx.x * blockDim.x + threadIdx.x;
  if (e >= nE) return;
  atomicAdd(&degTd[dst[e]], 1.0f);
  atomicAdd(&degBu[src[e]], 1.0f);
}

__global__ void k_dinv(float* __restrict__ d0, float* __restrict__ d1, int n) {
  int i = blockIdx.x * blockDim.x + threadIdx.x;
  if (i >= n) return;
  d0[i] = rsqrtf(d0[i] + 1.0f);
  d1[i] = rsqrtf(d1[i] + 1.0f);
}

__global__ void k_flags(const int* __restrict__ maskN, int* __restrict__ flags, int m) {
  int i = blockIdx.x * blockDim.x + threadIdx.x;
  if (i >= m) return;
  flags[maskN[i]] = 1;
}

// ---------------- fused WMMA GEMM + GCN epilogue ----------------
// For NW weight sets (1 or 2, sharing the same A):
//   lin[w]  = bf16(A) @ bf16(W[w])                       (needed by edge gather)
//   agg[w]  = lin[w] * dinv[w][row]^2 + bias[w][col]     (self-loop + bias seed)
// flags/token: masked rows of A replaced by the mask token (online encoder).
// reluA: relu applied to A on load (fuses relu(hidden) into the layer-2 GEMM).
template<int KD, int ND, int NW>
__global__ __launch_bounds__(256)
void gemm_wmma(const float* __restrict__ A,
               const int*   __restrict__ flags,
               const float* __restrict__ token,
               const float* __restrict__ W0,    const float* __restrict__ W1,
               const float* __restrict__ bias0, const float* __restrict__ bias1,
               const float* __restrict__ dinv0, const float* __restrict__ dinv1,
               float* __restrict__ lin0,  float* __restrict__ lin1,
               float* __restrict__ agg0,  float* __restrict__ agg1,
               int nrows, int reluA)
{
  constexpr int NT = ND / 16;          // 16-wide column tiles per weight set
  constexpr int KP = 40;               // padded K stride (bf16 elems), 80B rows
  __shared__ __bf16 wlds[NW * ND * KP];  // W tiles, transposed: [w][col][k]

  const int tid  = threadIdx.x;
  const int wave = tid >> 5;
  const int lane = tid & 31;
  const int m    = lane & 15;          // row within strip / column within tile
  const int half = lane >> 4;          // K-half select (this code's contiguous A/B k order)
  const int rowBase = blockIdx.x * 128 + wave * 16;
  const int row  = rowBase + m;
  const bool rowOK  = row < nrows;
  const bool useTok = rowOK && (flags != nullptr) && (flags[row] != 0);
  const float* arow = useTok ? token : (A + (size_t)row * KD);

  const float* Wt[NW];
  Wt[0] = W0;
  if constexpr (NW == 2) Wt[1] = W1;

  v8f zero = {};
  v8f acc[NW][NT];
#pragma unroll
  for (int w = 0; w < NW; ++w)
#pragma unroll
    for (int t = 0; t < NT; ++t) acc[w][t] = zero;

  for (int kb = 0; kb < KD; kb += 32) {
    // cooperative load of W tiles [kb..kb+32) x [0..ND), fp32 -> bf16, transposed
#pragma unroll
    for (int w = 0; w < NW; ++w) {
      for (int idx = tid; idx < 32 * ND; idx += 256) {
        int k = idx / ND, c = idx % ND;            // consecutive tid -> consecutive c
        wlds[w * ND * KP + c * KP + k] = (__bf16)Wt[w][(size_t)(kb + k) * ND + c];
      }
    }
    __syncthreads();

    // A fragment: 16x32 bf16; lanes 0-15: K 0..15; lanes 16-31: K 16..31
    v16bf af;
    {
      const int k0 = kb + half * 16;
      float va[16];
      if (rowOK) {
        float4 p0 = *(const float4*)(arow + k0);
        float4 p1 = *(const float4*)(arow + k0 + 4);
        float4 p2 = *(const float4*)(arow + k0 + 8);
        float4 p3 = *(const float4*)(arow + k0 + 12);
        va[0]=p0.x;  va[1]=p0.y;  va[2]=p0.z;  va[3]=p0.w;
        va[4]=p1.x;  va[5]=p1.y;  va[6]=p1.z;  va[7]=p1.w;
        va[8]=p2.x;  va[9]=p2.y;  va[10]=p2.z; va[11]=p2.w;
        va[12]=p3.x; va[13]=p3.y; va[14]=p3.z; va[15]=p3.w;
      } else {
#pragma unroll
        for (int j = 0; j < 16; ++j) va[j] = 0.0f;
      }
#pragma unroll
      for (int j = 0; j < 16; ++j) {
        float v = va[j];
        if (reluA) v = fmaxf(v, 0.0f);
        af[j] = (__bf16)v;
      }
    }

    // WMMAs; B frag: lane = column, lanes 0-15 hold K0-15, lanes 16-31 K16-31
#pragma unroll
    for (int w = 0; w < NW; ++w) {
#pragma unroll
      for (int t = 0; t < NT; ++t) {
        v16bf bf;
        const __bf16* bp = &wlds[w * ND * KP + (t * 16 + m) * KP + half * 16];
#pragma unroll
        for (int j = 0; j < 16; ++j) bf[j] = bp[j];
        acc[w][t] = __builtin_amdgcn_wmma_f32_16x16x32_bf16(
            false, af, false, bf, (short)0, acc[w][t], false, false);
      }
    }
    __syncthreads();
  }

  // epilogue: C VGPR v -> row rowBase+v (lanes 0-15) / +8 (lanes 16-31), col = t*16+m
#pragma unroll
  for (int w = 0; w < NW; ++w) {
    const float* dv = (w == 0) ? dinv0 : dinv1;
    const float* bs = (w == 0) ? bias0 : bias1;
    float* lo = (w == 0) ? lin0 : lin1;
    float* ao = (w == 0) ? agg0 : agg1;
    float di2[8];
#pragma unroll
    for (int v = 0; v < 8; ++v) {
      int r = rowBase + v + half * 8;
      float di = (r < nrows) ? dv[r] : 0.0f;
      di2[v] = di * di;
    }
#pragma unroll
    for (int t = 0; t < NT; ++t) {
      int c = t * 16 + m;
      float bval = bs[c];
#pragma unroll
      for (int v = 0; v < 8; ++v) {
        int r = rowBase + v + half * 8;
        if (r < nrows) {
          float val = acc[w][t][v];
          lo[(size_t)r * ND + c] = val;
          ao[(size_t)r * ND + c] = val * di2[v] + bval;
        }
      }
    }
  }
}

// ---------------- normalized edge scatter-add (segment_sum) ----------------
__global__ void k_edge_scatter(const float* __restrict__ lin,
                               const int* __restrict__ sidx, const int* __restrict__ didx,
                               const float* __restrict__ dinv, float* __restrict__ agg,
                               int nE, int F) {
  long long idx = (long long)blockIdx.x * blockDim.x + threadIdx.x;
  if (idx >= (long long)nE * F) return;
  int e = (int)(idx / F);
  int f = (int)(idx % F);
  int s = sidx[e], d = didx[e];
  float nrm = dinv[s] * dinv[d];
  atomicAdd(&agg[(size_t)d * F + f], lin[(size_t)s * F + f] * nrm);
}

// ---------------- global_add_pool ----------------
__global__ void k_pool(const float* __restrict__ h, const int* __restrict__ batch,
                       float* __restrict__ pool, int n, int F) {
  int idx = blockIdx.x * blockDim.x + threadIdx.x;
  if (idx >= n * F) return;
  int node = idx / F, f = idx % F;
  atomicAdd(&pool[(size_t)batch[node] * F + f], h[idx]);
}

// ---------------- scaled-cosine-error over concat([.,64],[.,64]) rows ----------------
__device__ __forceinline__ float waveReduceSum(float v) {
#pragma unroll
  for (int off = 16; off > 0; off >>= 1) v += __shfl_xor(v, off, 32);
  return v;
}

__global__ void k_sce(const float* __restrict__ a0, const float* __restrict__ a1,
                      const float* __restrict__ b0, const float* __restrict__ b1,
                      const int* __restrict__ idxMap, int count, float* __restrict__ accum) {
  int w    = (blockIdx.x * blockDim.x + threadIdx.x) >> 5;  // one wave per row
  int lane = threadIdx.x & 31;
  if (w >= count) return;
  int n = idxMap ? idxMap[w] : w;
  int f0 = lane * 4;                                         // 128 dims, 4 per lane
  const float* pa = (f0 < 64) ? (a0 + (size_t)n * 64 + f0) : (a1 + (size_t)n * 64 + f0 - 64);
  const float* pb = (f0 < 64) ? (b0 + (size_t)n * 64 + f0) : (b1 + (size_t)n * 64 + f0 - 64);
  float dot = 0.f, na = 0.f, nb = 0.f;
#pragma unroll
  for (int j = 0; j < 4; ++j) {
    float av = pa[j], bv = pb[j];
    dot += av * bv; na += av * av; nb += bv * bv;
  }
  dot = waveReduceSum(dot);
  na  = waveReduceSum(na);
  nb  = waveReduceSum(nb);
  if (lane == 0) {
    float denom = fmaxf(sqrtf(na), 1e-12f) * fmaxf(sqrtf(nb), 1e-12f);
    atomicAdd(accum, 1.0f - dot / denom);
  }
}

__global__ void k_finalize(const float* __restrict__ accum, float* __restrict__ out) {
  if (blockIdx.x == 0 && threadIdx.x == 0)
    out[0] = accum[0] / (float)N_MASK + accum[1] / (float)N_GRAPH;
}

// ---------------- host orchestration ----------------
extern "C" void kernel_launch(void* const* d_in, const int* in_sizes, int n_in,
                              void* d_out, int out_size, void* d_ws, size_t ws_size,
                              hipStream_t stream) {
  const float* x     = (const float*)d_in[0];
  const float* token = (const float*)d_in[1];
  const int*   ei    = (const int*)d_in[2];
  const int*   src   = ei;
  const int*   dst   = ei + N_EDGES;
  const int*   batch = (const int*)d_in[3];
  const int*   maskN = (const int*)d_in[4];

  // workspace carve-out
  char* p = (char*)d_ws;
  auto carve = [&](size_t bytes) -> void* {
    void* r = (void*)p;
    p += (bytes + 255) & ~(size_t)255;
    return r;
  };
  float* dinv_td = (float*)carve(sizeof(float) * N_NODES);
  float* dinv_bu = (float*)carve(sizeof(float) * N_NODES);
  int*   flags   = (int*)  carve(sizeof(int)   * N_NODES);
  float* linTd   = (float*)carve(sizeof(float) * (size_t)N_NODES * HID);  // lin1_td, later lin2_td
  float* linBu   = (float*)carve(sizeof(float) * (size_t)N_NODES * HID);  // lin1_bu, later lin2_bu
  float* aggTd   = (float*)carve(sizeof(float) * (size_t)N_NODES * HID);  // agg1_td -> hidden_td
  float* aggBu   = (float*)carve(sizeof(float) * (size_t)N_NODES * HID);  // agg1_bu -> hidden_bu
  float* hout[4];
  for (int b = 0; b < 4; ++b)
    hout[b] = (float*)carve(sizeof(float) * (size_t)N_NODES * OUTD);
  float* pools   = (float*)carve(sizeof(float) * 4 * N_GRAPH * OUTD);
  float* accum   = (float*)carve(sizeof(float) * 2);

  hipMemsetAsync(dinv_td, 0, sizeof(float) * N_NODES, stream);
  hipMemsetAsync(dinv_bu, 0, sizeof(float) * N_NODES, stream);
  hipMemsetAsync(flags,   0, sizeof(int)   * N_NODES, stream);
  hipMemsetAsync(pools,   0, sizeof(float) * 4 * N_GRAPH * OUTD, stream);
  hipMemsetAsync(accum,   0, sizeof(float) * 2, stream);

  const int T = 256;
  k_degree<<<(N_EDGES + T - 1) / T, T, 0, stream>>>(src, dst, dinv_td, dinv_bu, N_EDGES);
  k_dinv  <<<(N_NODES + T - 1) / T, T, 0, stream>>>(dinv_td, dinv_bu, N_NODES);
  k_flags <<<(N_MASK  + T - 1) / T, T, 0, stream>>>(maskN, flags, N_MASK);

  const int gemmGrid   = (N_NODES + 127) / 128;
  const int scat1Grid  = (int)(((long long)N_EDGES * HID  + T - 1) / T);
  const int scat2Grid  = (int)(((long long)N_EDGES * OUTD + T - 1) / T);
  const int poolGrid   = ((N_NODES * OUTD) + T - 1) / T;

  for (int enc = 0; enc < 2; ++enc) {
    const bool   online = (enc == 0);
    const float* W1td = (const float*)d_in[5 + 8 * enc];
    const float* b1td = (const float*)d_in[6 + 8 * enc];
    const float* W2td = (const float*)d_in[7 + 8 * enc];
    const float* b2td = (const float*)d_in[8 + 8 * enc];
    const float* W1bu = (const float*)d_in[9 + 8 * enc];
    const float* b1bu = (const float*)d_in[10 + 8 * enc];
    const float* W2bu = (const float*)d_in[11 + 8 * enc];
    const float* b2bu = (const float*)d_in[12 + 8 * enc];

    // layer 1, TD+BU fused (single pass over A, both W1s in LDS)
    gemm_wmma<IN_DIM, HID, 2><<<gemmGrid, T, 0, stream>>>(
        x, online ? flags : nullptr, online ? token : nullptr,
        W1td, W1bu, b1td, b1bu, dinv_td, dinv_bu,
        linTd, linBu, aggTd, aggBu, N_NODES, 0);
    k_edge_scatter<<<scat1Grid, T, 0, stream>>>(linTd, src, dst, dinv_td, aggTd, N_EDGES, HID);
    k_edge_scatter<<<scat1Grid, T, 0, stream>>>(linBu, dst, src, dinv_bu, aggBu, N_EDGES, HID);

    // layer 2, TD (relu fused into A-load; lin2 reuses linTd buffer)
    gemm_wmma<HID, OUTD, 1><<<gemmGrid, T, 0, stream>>>(
        aggTd, nullptr, nullptr, W2td, nullptr, b2td, nullptr, dinv_td, nullptr,
        linTd, nullptr, hout[2 * enc + 0], nullptr, N_NODES, 1);
    k_edge_scatter<<<scat2Grid, T, 0, stream>>>(linTd, src, dst, dinv_td,
                                                hout[2 * enc + 0], N_EDGES, OUTD);
    k_pool<<<poolGrid, T, 0, stream>>>(hout[2 * enc + 0], batch,
                                       pools + (size_t)(2 * enc + 0) * N_GRAPH * OUTD,
                                       N_NODES, OUTD);

    // layer 2, BU (edge direction flipped)
    gemm_wmma<HID, OUTD, 1><<<gemmGrid, T, 0, stream>>>(
        aggBu, nullptr, nullptr, W2bu, nullptr, b2bu, nullptr, dinv_bu, nullptr,
        linBu, nullptr, hout[2 * enc + 1], nullptr, N_NODES, 1);
    k_edge_scatter<<<scat2Grid, T, 0, stream>>>(linBu, dst, src, dinv_bu,
                                                hout[2 * enc + 1], N_EDGES, OUTD);
    k_pool<<<poolGrid, T, 0, stream>>>(hout[2 * enc + 1], batch,
                                       pools + (size_t)(2 * enc + 1) * N_GRAPH * OUTD,
                                       N_NODES, OUTD);
  }

  // node loss: h1 = cat(on_td, on_bu), h2 = cat(tgt_td, tgt_bu) at mask_nodes
  k_sce<<<(N_MASK * 32 + T - 1) / T, T, 0, stream>>>(
      hout[0], hout[1], hout[2], hout[3], maskN, N_MASK, accum + 0);
  // graph loss: gh = cat(bu_pool, td_pool) per encoder
  k_sce<<<(N_GRAPH * 32 + T - 1) / T, T, 0, stream>>>(
      pools + (size_t)1 * N_GRAPH * OUTD,   // on_bu pool
      pools + (size_t)0 * N_GRAPH * OUTD,   // on_td pool
      pools + (size_t)3 * N_GRAPH * OUTD,   // tgt_bu pool
      pools + (size_t)2 * N_GRAPH * OUTD,   // tgt_td pool
      nullptr, N_GRAPH, accum + 1);

  k_finalize<<<1, 1, 0, stream>>>(accum, (float*)d_out);
}
